// LSTM_54915451846935
// MI455X (gfx1250) — matrix operand order repaired and checked by reading, not compiled, not run on hardware
//
#include <hip/hip_runtime.h>
#include <hip/hip_bf16.h>

// ---------------------------------------------------------------------------
// BiLSTM classifier for MI455X (gfx1250, wave32, WMMA bf16 16x16x32).
//   V=50000 E=300 H=256 L=5 B=64 T_C=637 T_T=24
// Pipeline:
//   1) gather embeddings -> bf16, K padded 300->320
//   2) input projection GEMM (WMMA bf16, f32 acc) -> pre = x@wih^T + b
//   3) persistent-workgroup LSTM scan: h in LDS (bf16, 32KB), gates held in
//      WMMA accumulators (each wave owns i/f/g/o tiles of the same hidden
//      columns -> cell update fully in registers), c in registers
//   4) attention + softmax + fc + log_softmax (VALU, one block per batch)
// ---------------------------------------------------------------------------

typedef __attribute__((ext_vector_type(16))) __bf16 v16bf;
typedef __attribute__((ext_vector_type(8)))  __bf16 v8bf;
typedef __attribute__((ext_vector_type(8)))  float  v8f;

static constexpr int BB  = 64;     // batch
static constexpr int TC  = 637;    // content length
static constexpr int TT  = 24;     // title length
static constexpr int E   = 300;    // embed dim
static constexpr int EP  = 320;    // embed dim padded to mult of 32 (WMMA K)
static constexpr int H   = 256;    // hidden
static constexpr int G4  = 1024;   // 4*H gates
static constexpr int DH  = 512;    // 2*H bi-dir
static constexpr int L   = 5;

#define WMMA_BF16(a, b, c) \
  __builtin_amdgcn_wmma_f32_16x16x32_bf16(false, (a), false, (b), (short)0, (c), false, false)

// A fragment: 16x32 bf16 tile, rows = mbase..mbase+15, cols = kb..kb+31.
// Layout used here (contiguous K, matching load_b_frag): lane<16 -> M=lane, K{0..15};
//             lane>=16 -> M=lane-16, K{16..31}.
__device__ __forceinline__ v16bf load_a_frag(const __bf16* base, int stride,
                                             int mbase, int kb, int lane) {
  const int half = lane >> 4, ln = lane & 15;
  const __bf16* p = base + (size_t)(mbase + ln) * stride + kb + half * 16;
  v8bf lo = *(const v8bf*)p;
  v8bf hi = *(const v8bf*)(p + 8);
  return __builtin_shufflevector(lo, hi, 0,1,2,3,4,5,6,7,8,9,10,11,12,13,14,15);
}

// B fragment: 32x16 bf16 tile from W stored [N][Kpad] (row-major, K contiguous).
// Layout used here (contiguous K, matching load_a_frag): lanes 0-15 hold N=lane, K=kb..kb+15; lanes 16-31 hold K=kb+16..kb+31.
__device__ __forceinline__ v16bf load_b_frag(const __bf16* W, int stride,
                                             int nbase, int kb, int lane) {
  const int half = lane >> 4, ln = lane & 15;
  const __bf16* p = W + (size_t)(nbase + ln) * stride + kb + half * 16;
  v8bf lo = *(const v8bf*)p;
  v8bf hi = *(const v8bf*)(p + 8);
  return __builtin_shufflevector(lo, hi, 0,1,2,3,4,5,6,7,8,9,10,11,12,13,14,15);
}

__device__ __forceinline__ float sigf(float x) { return 1.f / (1.f + __expf(-x)); }
__device__ __forceinline__ float tanh_fast(float x) {
  x = fminf(fmaxf(x, -15.f), 15.f);
  float e = __expf(2.f * x);
  return (e - 1.f) / (e + 1.f);
}

// ---------------------------------------------------------------------------
// 1) Embedding gather -> bf16, pad E 300 -> 320
// grid = nrows, block = EP
__global__ void k_gather(const int* __restrict__ tok, const float* __restrict__ emb,
                         __bf16* __restrict__ dst) {
  const int row = blockIdx.x, c = threadIdx.x;
  const int t = tok[row];
  dst[(size_t)row * EP + c] = (c < E) ? (__bf16)emb[(size_t)t * E + c] : (__bf16)0.f;
}

// Weight convert f32 -> bf16, [rows][scols] -> [rows][dcols] zero-padded
// grid = rows, block = dcols
__global__ void k_convert(const float* __restrict__ src, __bf16* __restrict__ dst,
                          int scols, int dcols) {
  const int r = blockIdx.x, c = threadIdx.x;
  dst[(size_t)r * dcols + c] = (c < scols) ? (__bf16)src[(size_t)r * scols + c] : (__bf16)0.f;
}

// ---------------------------------------------------------------------------
// 2) Input projection: pre[M][1024] = X[M][Kpad] @ W[1024][Kpad]^T + bias
// block = 256 (8 waves); each wave computes a 16x64 output tile.
// grid = (M/16, 2)
__global__ void k_input_proj(const __bf16* __restrict__ X, const __bf16* __restrict__ W,
                             const float* __restrict__ bias, float* __restrict__ pre,
                             int M, int Kpad) {
  const int tid  = threadIdx.x;
  const int lane = tid & 31, wave = tid >> 5;
  const int half = lane >> 4, ln = lane & 15;
  const int mbase  = blockIdx.x * 16;
  const int nbase0 = (blockIdx.y * 8 + wave) * 64;

  v8f acc[4];
#pragma unroll
  for (int j = 0; j < 4; ++j) acc[j] = (v8f)0.f;

  for (int kb = 0; kb < Kpad; kb += 32) {
    v16bf a = load_a_frag(X, Kpad, mbase, kb, lane);
#pragma unroll
    for (int j = 0; j < 4; ++j) {
      v16bf b = load_b_frag(W, Kpad, nbase0 + j * 16, kb, lane);
      acc[j] = WMMA_BF16(a, b, acc[j]);
    }
  }
#pragma unroll
  for (int j = 0; j < 4; ++j) {
    const int n = nbase0 + j * 16 + ln;
    const float bn = bias[n];
#pragma unroll
    for (int r = 0; r < 8; ++r) {
      const int m = mbase + r + half * 8;
      pre[(size_t)m * G4 + n] = acc[j][r] + bn;
    }
  }
}

// ---------------------------------------------------------------------------
// 3) LSTM scan. One block per direction (blockIdx.x: 0=fwd, 1=bwd), 512 thr.
// LDS: h bf16 [64][256] = 32KB only. Wave (m_tile, ngrp) owns hidden-column
// tiles jh = ngrp*4..ngrp*4+3; for each it computes the i/f/g/o gate tiles
// (N-tiles jh, jh+16, jh+32, jh+48) so the cell update stays in registers.
__global__ void k_lstm_scan(const float* __restrict__ pre_f, const float* __restrict__ pre_b,
                            const __bf16* __restrict__ whh_f, const __bf16* __restrict__ whh_b,
                            float* __restrict__ out, int T) {
  __shared__ __bf16 hbuf[BB * H];  // 32KB

  const int rev  = blockIdx.x;
  const float*  pre = rev ? pre_b : pre_f;
  const __bf16* whh = rev ? whh_b : whh_f;
  const int coff = rev * H;

  const int tid  = threadIdx.x;          // 512 threads = 16 waves
  const int lane = tid & 31, wave = tid >> 5;
  const int half = lane >> 4, ln = lane & 15;
  const int mbase = (wave & 3) * 16;     // 4 M-tiles cover B=64
  const int ngrp  = wave >> 2;           // 4 groups x 4 h-column tiles = H

  float c_reg[32];
#pragma unroll
  for (int i = 0; i < 32; ++i) c_reg[i] = 0.f;
  for (int i = tid; i < BB * H; i += 512) hbuf[i] = (__bf16)0.f;
  __syncthreads();

  for (int s = 0; s < T; ++s) {
    const int t = rev ? (T - 1 - s) : s;

    // read h fragments for this step (all reads complete before any write)
    v16bf afr[8];
#pragma unroll
    for (int kk = 0; kk < 8; ++kk)
      afr[kk] = load_a_frag(hbuf, H, mbase, kk * 32, lane);  // ds_load_b128
    __syncthreads();

    // prefetch next step's pre rows for this wave (hides strided f32 loads)
    if (s + 1 < T) {
      const int tn = rev ? (t - 1) : (t + 1);
#pragma unroll
      for (int r = 0; r < 8; ++r) {
        const int m = mbase + r + half * 8;
        __builtin_prefetch(&pre[((size_t)m * T + tn) * G4 + ngrp * 64 + ln], 0, 0);
      }
    }

#pragma unroll
    for (int jl = 0; jl < 4; ++jl) {
      const int jh = ngrp * 4 + jl;      // h-column tile (0..15)
      const int nb = jh * 16;            // column base within H
      // init accumulators with pre-projection (includes bias)
      v8f ai, af, ag, ao;
#pragma unroll
      for (int r = 0; r < 8; ++r) {
        const int m = mbase + r + half * 8;
        const size_t ro = ((size_t)m * T + t) * G4;
        ai[r] = pre[ro + nb + ln];
        af[r] = pre[ro + H + nb + ln];
        ag[r] = pre[ro + 2 * H + nb + ln];
        ao[r] = pre[ro + 3 * H + nb + ln];
      }
      // gates += h @ whh^T for the four gate tiles of these columns
#pragma unroll
      for (int kk = 0; kk < 8; ++kk) {
        const int kb = kk * 32;
        ai = WMMA_BF16(afr[kk], load_b_frag(whh, H, nb, kb, lane), ai);
        af = WMMA_BF16(afr[kk], load_b_frag(whh, H, H + nb, kb, lane), af);
        ag = WMMA_BF16(afr[kk], load_b_frag(whh, H, 2 * H + nb, kb, lane), ag);
        ao = WMMA_BF16(afr[kk], load_b_frag(whh, H, 3 * H + nb, kb, lane), ao);
      }
      // LSTM cell update entirely in registers
#pragma unroll
      for (int r = 0; r < 8; ++r) {
        const int m = mbase + r + half * 8;
        const int n = nb + ln;
        const float c = sigf(af[r]) * c_reg[jl * 8 + r] + sigf(ai[r]) * tanh_fast(ag[r]);
        c_reg[jl * 8 + r] = c;
        const float h = sigf(ao[r]) * tanh_fast(c);
        hbuf[m * H + n] = (__bf16)h;
        out[((size_t)m * T + t) * DH + coff + n] = h;
      }
    }
    __syncthreads();  // h fully written before next step's reads
  }
}

// ---------------------------------------------------------------------------
// 4) attention + softmax + fc + log_softmax. grid = 64 (per batch), block = 256
__global__ void k_attn(const float* __restrict__ trep, const float* __restrict__ crep,
                       const float* __restrict__ att_w, const float* __restrict__ fc_w,
                       const float* __restrict__ fc_b, float* __restrict__ out) {
  __shared__ float tl[DH];
  __shared__ float v[DH];
  __shared__ float a[TC + 3];
  __shared__ float red[256];
  __shared__ float logits[8];

  const int b = blockIdx.x, tid = threadIdx.x;

  for (int j = tid; j < DH; j += 256) tl[j] = trep[((size_t)b * TT + (TT - 1)) * DH + j];
  __syncthreads();

  for (int k = tid; k < DH; k += 256) {
    float s = 0.f;
    for (int j = 0; j < DH; ++j) s += tl[j] * att_w[(size_t)j * DH + k];
    v[k] = s;
  }
  __syncthreads();

  float lmax = -1e30f;
  for (int t = tid; t < TC; t += 256) {
    const float* cr = crep + ((size_t)b * TC + t) * DH;
    float s = 0.f;
    for (int k = 0; k < DH; ++k) s += v[k] * cr[k];
    a[t] = s;
    lmax = fmaxf(lmax, s);
  }
  red[tid] = lmax; __syncthreads();
  for (int o = 128; o > 0; o >>= 1) {
    if (tid < o) red[tid] = fmaxf(red[tid], red[tid + o]);
    __syncthreads();
  }
  const float mx = red[0]; __syncthreads();

  float lsum = 0.f;
  for (int t = tid; t < TC; t += 256) { float e = __expf(a[t] - mx); a[t] = e; lsum += e; }
  red[tid] = lsum; __syncthreads();
  for (int o = 128; o > 0; o >>= 1) {
    if (tid < o) red[tid] += red[tid + o];
    __syncthreads();
  }
  const float inv = 1.f / red[0]; __syncthreads();

  if (tid < L) {
    float s = 0.f;
    for (int t = 0; t < TC; ++t) s += a[t] * inv * fc_w[(size_t)tid * TC + t];
    logits[tid] = s + fc_b[tid];
  }
  __syncthreads();
  if (tid == 0) {
    float m2 = logits[0];
    for (int l = 1; l < L; ++l) m2 = fmaxf(m2, logits[l]);
    float se = 0.f;
    for (int l = 0; l < L; ++l) se += __expf(logits[l] - m2);
    const float lse = m2 + __logf(se);
    for (int l = 0; l < L; ++l) out[b * L + l] = logits[l] - lse;
  }
}

// ---------------------------------------------------------------------------
extern "C" void kernel_launch(void* const* d_in, const int* in_sizes, int n_in,
                              void* d_out, int out_size, void* d_ws, size_t ws_size,
                              hipStream_t stream) {
  (void)in_sizes; (void)n_in; (void)out_size; (void)ws_size;
  const int*   content = (const int*)d_in[0];
  const int*   title   = (const int*)d_in[1];
  const float* embed   = (const float*)d_in[2];
  const float* t_wih_f = (const float*)d_in[3];
  const float* t_whh_f = (const float*)d_in[4];
  const float* t_b_f   = (const float*)d_in[5];
  const float* t_wih_b = (const float*)d_in[6];
  const float* t_whh_b = (const float*)d_in[7];
  const float* t_b_b   = (const float*)d_in[8];
  const float* c_wih_f = (const float*)d_in[9];
  const float* c_whh_f = (const float*)d_in[10];
  const float* c_b_f   = (const float*)d_in[11];
  const float* c_wih_b = (const float*)d_in[12];
  const float* c_whh_b = (const float*)d_in[13];
  const float* c_b_b   = (const float*)d_in[14];
  const float* att_w   = (const float*)d_in[15];
  const float* fc_w    = (const float*)d_in[16];
  const float* fc_b    = (const float*)d_in[17];
  float* out = (float*)d_out;

  char* ws = (char*)d_ws;
  size_t off = 0;
  auto alloc = [&](size_t bytes) -> char* {
    char* p = ws + off;
    off = (off + bytes + 255) & ~(size_t)255;
    return p;
  };

  const int MC = BB * TC;   // 40768, mult of 16
  const int MT = BB * TT;   // 1536,  mult of 16

  __bf16* xc = (__bf16*)alloc((size_t)MC * EP * 2);
  __bf16* xt = (__bf16*)alloc((size_t)MT * EP * 2);
  __bf16* wih_tf = (__bf16*)alloc((size_t)G4 * EP * 2);
  __bf16* wih_tb = (__bf16*)alloc((size_t)G4 * EP * 2);
  __bf16* wih_cf = (__bf16*)alloc((size_t)G4 * EP * 2);
  __bf16* wih_cb = (__bf16*)alloc((size_t)G4 * EP * 2);
  __bf16* whh_tf = (__bf16*)alloc((size_t)G4 * H * 2);
  __bf16* whh_tb = (__bf16*)alloc((size_t)G4 * H * 2);
  __bf16* whh_cf = (__bf16*)alloc((size_t)G4 * H * 2);
  __bf16* whh_cb = (__bf16*)alloc((size_t)G4 * H * 2);
  float* pre_tf = (float*)alloc((size_t)MT * G4 * 4);
  float* pre_tb = (float*)alloc((size_t)MT * G4 * 4);
  float* pre_cf = (float*)alloc((size_t)MC * G4 * 4);
  float* pre_cb = (float*)alloc((size_t)MC * G4 * 4);
  float* trep = (float*)alloc((size_t)BB * TT * DH * 4);
  float* crep = (float*)alloc((size_t)BB * TC * DH * 4);

  // 0) weight conversion to bf16 [N][Kpad]
  k_convert<<<G4, EP, 0, stream>>>(t_wih_f, wih_tf, E, EP);
  k_convert<<<G4, EP, 0, stream>>>(t_wih_b, wih_tb, E, EP);
  k_convert<<<G4, EP, 0, stream>>>(c_wih_f, wih_cf, E, EP);
  k_convert<<<G4, EP, 0, stream>>>(c_wih_b, wih_cb, E, EP);
  k_convert<<<G4, H, 0, stream>>>(t_whh_f, whh_tf, H, H);
  k_convert<<<G4, H, 0, stream>>>(t_whh_b, whh_tb, H, H);
  k_convert<<<G4, H, 0, stream>>>(c_whh_f, whh_cf, H, H);
  k_convert<<<G4, H, 0, stream>>>(c_whh_b, whh_cb, H, H);

  // 1) embedding gather
  k_gather<<<MC, EP, 0, stream>>>(content, embed, xc);
  k_gather<<<MT, EP, 0, stream>>>(title, embed, xt);

  // 2) input projections (WMMA GEMM)
  k_input_proj<<<dim3(MT / 16, 2), 256, 0, stream>>>(xt, wih_tf, t_b_f, pre_tf, MT, EP);
  k_input_proj<<<dim3(MT / 16, 2), 256, 0, stream>>>(xt, wih_tb, t_b_b, pre_tb, MT, EP);
  k_input_proj<<<dim3(MC / 16, 2), 256, 0, stream>>>(xc, wih_cf, c_b_f, pre_cf, MC, EP);
  k_input_proj<<<dim3(MC / 16, 2), 256, 0, stream>>>(xc, wih_cb, c_b_b, pre_cb, MC, EP);

  // 3) LSTM scans (fwd+bwd as 2 blocks each), only 32KB LDS now
  k_lstm_scan<<<2, 512, 0, stream>>>(pre_tf, pre_tb, whh_tf, whh_tb, trep, TT);
  k_lstm_scan<<<2, 512, 0, stream>>>(pre_cf, pre_cb, whh_cf, whh_cb, crep, TC);

  // 4) attention + softmax + fc + log_softmax
  k_attn<<<BB, 256, 0, stream>>>(trep, crep, att_w, fc_w, fc_b, out);
}
